// Head_24283745092034
// MI455X (gfx1250) — matrix operand order repaired and checked by reading, not compiled, not run on hardware
//
#include <hip/hip_runtime.h>

typedef __attribute__((ext_vector_type(16))) __bf16 v16bf;
typedef __attribute__((ext_vector_type(8)))  __bf16 v8bf;
typedef __attribute__((ext_vector_type(8)))  float  v8f;
typedef __attribute__((ext_vector_type(4)))  float  v4f;

#define B_DIM 8
#define T_DIM 2048
#define C_DIM 1024
#define H_DIM 64
// (1/sqrt(H)) * log2(e): fold softmax scale + exp->exp2 conversion into q
#define QSCALE 0.18033688011112042f

// Native conversion: lowers to v_cvt_pk_bf16_f32 (RNE) instead of bit-twiddling
static __device__ __forceinline__ __bf16 f2bf(float f) { return (__bf16)f; }

// ---------------------------------------------------------------------------
// Kernel 0: convert Wq/Wk/Wv (fp32 [1024][64]) to bf16 in WMMA-B-fragment
// friendly layout: Wt[w][c][kb][n][e] = W[kb*16+e][c*16+n]  (w: 0=q,1=k,2=v)
// so a lane's 16 K-values for one column are contiguous (32B load).
// Wq is pre-scaled by QSCALE.
// ---------------------------------------------------------------------------
__global__ __launch_bounds__(256) void wconvert(const float* __restrict__ Wk,
                                                const float* __restrict__ Wq,
                                                const float* __restrict__ Wv,
                                                __bf16* __restrict__ Wt) {
  int idx = blockIdx.x * 256 + threadIdx.x;    // 3*1024*64 = 196608 threads
  int w   = idx >> 16;
  int rem = idx & 65535;
  int k = rem >> 6;
  int n = rem & 63;
  const float* src = (w == 0) ? Wq : (w == 1) ? Wk : Wv;
  float v = src[k * H_DIM + n];
  if (w == 0) v *= QSCALE;
  int c = n >> 4, nn = n & 15, kbk = k >> 4, e = k & 15;
  Wt[(size_t)w * 65536 + (((c * 64 + kbk) * 16 + nn) * 16) + e] = f2bf(v);
}

// ---------------------------------------------------------------------------
// Kernel 1: fused q/k/v projection via bf16 WMMA.  grid = 1024 row tiles,
// 4 waves/block, wave = one 16-column tile.  Each wave loads the shared x
// A-fragment ONCE per K-step and issues 3 WMMAs (q, k, v).  Writes:
//   qb,kb: bf16 [B*T][64] row-major;   vt: bf16 [B][64][T] (transposed)
// ---------------------------------------------------------------------------
__global__ __launch_bounds__(128) void qkv_proj(const float* __restrict__ x,
                                                const __bf16* __restrict__ Wt,
                                                __bf16* __restrict__ qb,
                                                __bf16* __restrict__ kbuf,
                                                __bf16* __restrict__ vt) {
  int lane = threadIdx.x & 31;
  int wave = threadIdx.x >> 5;                 // column tile c = 0..3
  int rt0  = blockIdx.x * 16;                  // output row tile (of 16384)
  int grp  = lane >> 4, lm = lane & 15;

  const __bf16* wq = Wt + (size_t)wave * 64 * 256;
  const __bf16* wk = wq + 65536;
  const __bf16* wv = wk + 65536;
  const float*  arow = x + (size_t)(rt0 + lm) * C_DIM;

  v8f aq = {}, ak = {}, av = {};
  for (int k0 = 0; k0 < C_DIM; k0 += 32) {
    // A fragment: lane lm holds row rt0+lm; elems 0..15 -> K=k0+grp*16+e
    v4f a0 = *(const v4f*)(arow + k0 + grp * 16);
    v4f a1 = *(const v4f*)(arow + k0 + grp * 16 + 4);
    v4f a2 = *(const v4f*)(arow + k0 + grp * 16 + 8);
    v4f a3 = *(const v4f*)(arow + k0 + grp * 16 + 12);
    v16bf A;
    for (int i = 0; i < 4; ++i) {
      A[i]      = f2bf(a0[i]);
      A[4 + i]  = f2bf(a1[i]);
      A[8 + i]  = f2bf(a2[i]);
      A[12 + i] = f2bf(a3[i]);
    }
    size_t boff = (size_t)((k0 >> 4) + grp) * 256 + lm * 16;
    v16bf Bq = *(const v16bf*)(wq + boff);
    v16bf Bk = *(const v16bf*)(wk + boff);
    v16bf Bv = *(const v16bf*)(wv + boff);
    aq = __builtin_amdgcn_wmma_f32_16x16x32_bf16(false, A, false, Bq, (short)0, aq, false, false);
    ak = __builtin_amdgcn_wmma_f32_16x16x32_bf16(false, A, false, Bk, (short)0, ak, false, false);
    av = __builtin_amdgcn_wmma_f32_16x16x32_bf16(false, A, false, Bv, (short)0, av, false, false);
  }

  // C tiles: lane lm, half grp holds rows rt0+8*grp+e, col wave*16+lm
  int h = wave * 16 + lm;
  for (int e = 0; e < 8; ++e) {
    int R = rt0 + 8 * grp + e;
    qb[(size_t)R * H_DIM + h]   = f2bf(aq[e]);
    kbuf[(size_t)R * H_DIM + h] = f2bf(ak[e]);
    int bb = R >> 11, t = R & (T_DIM - 1);
    vt[((size_t)bb * H_DIM + h) * T_DIM + t] = f2bf(av[e]);
  }
}

// ---------------------------------------------------------------------------
// Kernel 2: causal flash attention.  grid=(32 q-tiles, 8 batches), 4 waves,
// each wave owns 16 query rows, streams keys 32 at a time.
// ---------------------------------------------------------------------------
__global__ __launch_bounds__(128) void attn(const __bf16* __restrict__ qb,
                                            const __bf16* __restrict__ kbuf,
                                            const __bf16* __restrict__ vt,
                                            float* __restrict__ out) {
  __shared__ __bf16 plds[4][16][32];           // per-wave P staging (C->A layout)

  int lane = threadIdx.x & 31;
  int wave = threadIdx.x >> 5;
  int b    = blockIdx.y;
  int qt0  = blockIdx.x * 64 + wave * 16;
  int grp  = lane >> 4, lm = lane & 15;

  // Q A-fragments for the two K-steps (h 0..31 and 32..63), loaded once
  const __bf16* qrow = qb + ((size_t)b * T_DIM + qt0 + lm) * H_DIM;
  v16bf Aq[2];
  for (int ks = 0; ks < 2; ++ks) {
    v8bf lo = *(const v8bf*)(qrow + ks * 32 + grp * 16);
    v8bf hi = *(const v8bf*)(qrow + ks * 32 + grp * 16 + 8);
    for (int i = 0; i < 8; ++i) { Aq[ks][i] = lo[i]; Aq[ks][8 + i] = hi[i]; }
  }

  v8f o0 = {}, o1 = {}, o2 = {}, o3 = {};
  float m[8], lsum[8];
  for (int e = 0; e < 8; ++e) { m[e] = -INFINITY; lsum[e] = 0.0f; }

  const __bf16* kbase = kbuf + (size_t)b * T_DIM * H_DIM;
  const __bf16* vbase = vt + (size_t)b * H_DIM * T_DIM;

  int qmax = qt0 + 15;
  for (int s0 = 0; s0 <= qmax; s0 += 32) {
    // Prefetch next key/value block while we compute on this one
    if (s0 + 32 <= qmax) {
      __builtin_prefetch(kbase + (size_t)(s0 + 32 + lm) * H_DIM, 0, 1);
      __builtin_prefetch(vbase + (size_t)lm * T_DIM + s0 + 32, 0, 1);
    }

    // --- S = Q K^T for keys [s0, s0+31], two 16-key tiles ---
    v8f S0 = {}, S1 = {};
    for (int ks = 0; ks < 2; ++ks) {
      v16bf B0 = *(const v16bf*)(kbase + (size_t)(s0 + lm) * H_DIM + ks * 32 + grp * 16);
      v16bf B1 = *(const v16bf*)(kbase + (size_t)(s0 + 16 + lm) * H_DIM + ks * 32 + grp * 16);
      S0 = __builtin_amdgcn_wmma_f32_16x16x32_bf16(false, Aq[ks], false, B0,
                                                   (short)0, S0, false, false);
      S1 = __builtin_amdgcn_wmma_f32_16x16x32_bf16(false, Aq[ks], false, B1,
                                                   (short)0, S1, false, false);
    }

    // --- causal mask + online softmax (per element slot = one query row) ---
    int key0 = s0 + lm, key1 = s0 + 16 + lm;
    float pr0[8], pr1[8], corr[8];
    for (int e = 0; e < 8; ++e) {
      int qg = qt0 + e + 8 * grp;
      float s0v = (key0 <= qg) ? S0[e] : -INFINITY;
      float s1v = (key1 <= qg) ? S1[e] : -INFINITY;
      float t = fmaxf(s0v, s1v);
      for (int msk = 1; msk <= 8; msk <<= 1)
        t = fmaxf(t, __shfl_xor(t, msk, 32));
      float mn = fmaxf(m[e], t);
      float c  = __builtin_exp2f(m[e] - mn);
      m[e] = mn;
      corr[e] = c;
      float p0 = __builtin_exp2f(s0v - mn);
      float p1 = __builtin_exp2f(s1v - mn);
      float rs = p0 + p1;
      for (int msk = 1; msk <= 8; msk <<= 1)
        rs += __shfl_xor(rs, msk, 32);
      lsum[e] = lsum[e] * c + rs;
      pr0[e] = p0; pr1[e] = p1;
    }
    for (int e = 0; e < 8; ++e) {
      o0[e] *= corr[e]; o1[e] *= corr[e]; o2[e] *= corr[e]; o3[e] *= corr[e];
    }

    // --- stage P (C layout) into LDS, reload in A layout ---
    for (int e = 0; e < 8; ++e) {
      plds[wave][e + 8 * grp][lm]      = f2bf(pr0[e]);
      plds[wave][e + 8 * grp][16 + lm] = f2bf(pr1[e]);
    }
    asm volatile("s_wait_dscnt 0" ::: "memory");
    v16bf Ap;
    {
      v8bf lo = *(const v8bf*)&plds[wave][lm][grp * 16];
      v8bf hi = *(const v8bf*)&plds[wave][lm][grp * 16 + 8];
      for (int i = 0; i < 8; ++i) { Ap[i] = lo[i]; Ap[8 + i] = hi[i]; }
    }

    // --- O += P * V; V^T layout makes each B fragment a contiguous load ---
    v16bf Bv0 = *(const v16bf*)(vbase + (size_t)(0  + lm) * T_DIM + s0 + grp * 16);
    v16bf Bv1 = *(const v16bf*)(vbase + (size_t)(16 + lm) * T_DIM + s0 + grp * 16);
    v16bf Bv2 = *(const v16bf*)(vbase + (size_t)(32 + lm) * T_DIM + s0 + grp * 16);
    v16bf Bv3 = *(const v16bf*)(vbase + (size_t)(48 + lm) * T_DIM + s0 + grp * 16);
    o0 = __builtin_amdgcn_wmma_f32_16x16x32_bf16(false, Ap, false, Bv0, (short)0, o0, false, false);
    o1 = __builtin_amdgcn_wmma_f32_16x16x32_bf16(false, Ap, false, Bv1, (short)0, o1, false, false);
    o2 = __builtin_amdgcn_wmma_f32_16x16x32_bf16(false, Ap, false, Bv2, (short)0, o2, false, false);
    o3 = __builtin_amdgcn_wmma_f32_16x16x32_bf16(false, Ap, false, Bv3, (short)0, o3, false, false);
  }

  // --- epilogue: divide by row sums, write fp32 output ---
  float* orow = out + (size_t)b * T_DIM * H_DIM;
  for (int e = 0; e < 8; ++e) {
    float inv = 1.0f / lsum[e];
    size_t r = (size_t)(qt0 + e + 8 * grp) * H_DIM;
    orow[r + lm]      = o0[e] * inv;
    orow[r + 16 + lm] = o1[e] * inv;
    orow[r + 32 + lm] = o2[e] * inv;
    orow[r + 48 + lm] = o3[e] * inv;
  }
}

// ---------------------------------------------------------------------------
extern "C" void kernel_launch(void* const* d_in, const int* in_sizes, int n_in,
                              void* d_out, int out_size, void* d_ws, size_t ws_size,
                              hipStream_t stream) {
  const float* x  = (const float*)d_in[0];
  const float* Wk = (const float*)d_in[1];
  const float* Wq = (const float*)d_in[2];
  const float* Wv = (const float*)d_in[3];
  float* out = (float*)d_out;

  char* base = (char*)d_ws;
  __bf16* Wt = (__bf16*)base;                                  // 384 KB
  __bf16* qb = (__bf16*)(base + 3 * 65536 * 2);                // 2 MB
  __bf16* kb = qb + (size_t)B_DIM * T_DIM * H_DIM;             // 2 MB
  __bf16* vt = kb + (size_t)B_DIM * T_DIM * H_DIM;             // 2 MB

  hipLaunchKernelGGL(wconvert, dim3(768), dim3(256), 0, stream, Wk, Wq, Wv, Wt);
  hipLaunchKernelGGL(qkv_proj, dim3(1024), dim3(128), 0, stream, x, Wt, qb, kb, vt);
  hipLaunchKernelGGL(attn, dim3(32, 8), dim3(128), 0, stream, qb, kb, vt, out);
}
